// SparsePatchSelector_32762010534293
// MI455X (gfx1250) — matrix operand order repaired and checked by reading, not compiled, not run on hardware
//
#include <hip/hip_runtime.h>
#include <hip/hip_bf16.h>

#define B_   32
#define C_   256
#define H_   112
#define W_   112
#define HW_  (H_ * W_)        // 12544
#define P_   8
#define NH_  14
#define NP_  196
#define K_   64
#define D_   256
#define CPP_ (C_ * P_ * P_)   // 16384
#define NSTEP (CPP_ / 32)     // 512 K-steps

typedef _Float16 v8h  __attribute__((ext_vector_type(8)));
typedef _Float16 v16h __attribute__((ext_vector_type(16)));
typedef float    v8f  __attribute__((ext_vector_type(8)));
typedef float    f32x4 __attribute__((ext_vector_type(4)));
typedef unsigned int u32x4 __attribute__((ext_vector_type(4)));
typedef int      i32x4 __attribute__((ext_vector_type(4)));
typedef int      i32x8 __attribute__((ext_vector_type(8)));

#define BSTR 40   // padded LDS row stride in halves (80 B): conflict-free b128
                  // fragment reads, and reproduced by TDM pad (16 dw + 4 dw)

// ---------------------------------------------------------------------------
// Kernel 1: saliency (1x1 conv over C) + per-block sigmoid partial sums.
// ---------------------------------------------------------------------------
__global__ void __launch_bounds__(256)
sal_kernel(const float* __restrict__ deep,
           const float* __restrict__ sal_w,
           const float* __restrict__ sal_b,
           float* __restrict__ ws_sal,
           float* __restrict__ ws_blocksum) {
  const int blk = blockIdx.x;
  const int b   = blk / 49;
  const int pix = (blk % 49) * 256 + threadIdx.x;   // [0, 12544)

  const float* dp = deep + (size_t)b * (C_ * HW_) + pix;
  float s = 0.f;
#pragma unroll 8
  for (int c = 0; c < C_; ++c)
    s += dp[(size_t)c * HW_] * sal_w[c];            // sal_w[c] wave-uniform -> s_load
  s += sal_b[0];

  ws_sal[(size_t)b * HW_ + pix] = s;

  __shared__ float red[256];
  red[threadIdx.x] = 1.f / (1.f + __expf(-s));
  __syncthreads();
  for (int off = 128; off > 0; off >>= 1) {
    if (threadIdx.x < off) red[threadIdx.x] += red[threadIdx.x + off];
    __syncthreads();
  }
  if (threadIdx.x == 0) ws_blocksum[blk] = red[0];
}

// ---------------------------------------------------------------------------
// Kernel 2: 8x8 avg-pool -> sal_flat [B,196]; ch_attn [B].
// ---------------------------------------------------------------------------
__global__ void __launch_bounds__(256)
pool_kernel(const float* __restrict__ ws_sal,
            const float* __restrict__ ws_blocksum,
            float* __restrict__ ws_salflat,
            float* __restrict__ out_chattn) {
  const int b = blockIdx.x;
  const int t = threadIdx.x;
  if (t < NP_) {
    const int ph = t / NH_, pw = t % NH_;
    const float* sp = ws_sal + (size_t)b * HW_ + (ph * P_) * W_ + pw * P_;
    float s = 0.f;
#pragma unroll
    for (int py = 0; py < P_; ++py)
#pragma unroll
      for (int px = 0; px < P_; ++px)
        s += sp[py * W_ + px];
    ws_salflat[b * NP_ + t] = s * (1.f / 64.f);
  }
  if (t == 0) {
    float s = 0.f;
    for (int i = 0; i < 49; ++i) s += ws_blocksum[b * 49 + i];
    out_chattn[b] = s * (1.f / (float)HW_);
  }
}

// ---------------------------------------------------------------------------
// Kernel 3: top-64 of 196 via bitonic sort of 256 (val,idx) pairs in LDS.
// ---------------------------------------------------------------------------
__device__ __forceinline__ bool before_desc(float va, int ia, float vb, int ib) {
  return (va > vb) || (va == vb && ia < ib);
}

__global__ void __launch_bounds__(256)
topk_kernel(const float* __restrict__ ws_salflat,
            int* __restrict__ ws_topidx,
            float* __restrict__ out_idx) {
  const int b = blockIdx.x;
  const int t = threadIdx.x;
  __shared__ float sv[256];
  __shared__ int   si[256];
  sv[t] = (t < NP_) ? ws_salflat[b * NP_ + t] : -3.4e38f;
  si[t] = t;
  __syncthreads();

  for (int k = 2; k <= 256; k <<= 1) {
    for (int j = k >> 1; j > 0; j >>= 1) {
      const int ixj = t ^ j;
      if (ixj > t) {
        const float vi = sv[t], vj = sv[ixj];
        const int   ii = si[t], ij = si[ixj];
        const bool dirDesc = ((t & k) == 0);
        const bool doswap = dirDesc ? before_desc(vj, ij, vi, ii)
                                    : before_desc(vi, ii, vj, ij);
        if (doswap) { sv[t] = vj; si[t] = ij; sv[ixj] = vi; si[ixj] = ii; }
      }
      __syncthreads();
    }
  }
  if (t < K_) {
    const int idx = si[t];
    ws_topidx[b * K_ + t] = idx;
    out_idx[b * K_ + t]   = (float)idx;
  }
}

// ---------------------------------------------------------------------------
// Kernel 3.5: transpose-convert proj_w [k][n] f32 -> projh [n][k] f16.
// ---------------------------------------------------------------------------
__global__ void __launch_bounds__(256)
convert_kernel(const float* __restrict__ proj_w, _Float16* __restrict__ projh) {
  __shared__ float tile[32][33];
  const int k0 = blockIdx.x * 32;
  const int n0 = blockIdx.y * 32;
  const int tx = threadIdx.x & 31;
  const int ty = threadIdx.x >> 5;
#pragma unroll
  for (int i = 0; i < 4; ++i)
    tile[ty + i * 8][tx] = proj_w[(size_t)(k0 + ty + i * 8) * D_ + (n0 + tx)];
  __syncthreads();
#pragma unroll
  for (int i = 0; i < 4; ++i)
    projh[(size_t)(n0 + ty + i * 8) * CPP_ + (k0 + tx)] =
        (_Float16)tile[tx][ty + i * 8];
}

// ---------------------------------------------------------------------------
// TDM: async 2D tile load  projh[n][k0..k0+31] f16 -> LDS, pad 16dw+4dw
// => 40-half row stride.  D# per CDNA5 ISA.
// ---------------------------------------------------------------------------
__device__ __forceinline__ void tdm_load_b(unsigned lds_off, const _Float16* gsrc) {
  const unsigned long long ga = (unsigned long long)(uintptr_t)gsrc;
  u32x4 g0;
  g0[0] = 1u;                                        // count=1 valid user D#
  g0[1] = lds_off;                                   // lds_addr (bytes)
  g0[2] = (unsigned)(ga & 0xFFFFFFFFu);              // global_addr[31:0]
  g0[3] = (unsigned)((ga >> 32) & 0x01FFFFFFu)       // global_addr[56:32]
        | (2u << 30);                                // type = 2 ("image")
  i32x8 g1;
  g1[0] = (1 << 16)                                  // data_size = 2 bytes
        | (1 << 20)                                  // pad_enable
        | (3 << 22)                                  // pad_interval: 16 dwords
        | (3 << 25);                                 // pad_amount:   4 dwords
  g1[1] = (int)(((unsigned)CPP_ & 0xFFFFu) << 16);   // tensor_dim0[15:0]
  g1[2] = (int)(((unsigned)CPP_ >> 16) & 0xFFFFu)    // tensor_dim0[31:16]
        | (int)(((unsigned)D_ & 0xFFFFu) << 16);     // tensor_dim1[15:0]
  g1[3] = (int)(32u << 16);                          // tile_dim0 = 32
  g1[4] = (int)D_;                                   // tile_dim1 = 256
  g1[5] = (int)CPP_;                                 // tensor_dim0_stride
  g1[6] = 0;
  g1[7] = 0;
  const i32x4 z4 = {0, 0, 0, 0};
#if defined(__clang_major__) && (__clang_major__ >= 23)
  const i32x8 z8 = {0, 0, 0, 0, 0, 0, 0, 0};
  __builtin_amdgcn_tensor_load_to_lds(g0, g1, z4, z4, z8, 0);
#else
  __builtin_amdgcn_tensor_load_to_lds(g0, g1, z4, z4, 0);
#endif
}

// ---------------------------------------------------------------------------
// Kernel 4: gathered-patch GEMM, M=2048 N=256 K=16384, f16 WMMA f32 acc.
// Double-buffered LDS, B via TDM (TENSORcnt), A via b128 gather,
// ONE barrier per K-step.
// ---------------------------------------------------------------------------
__global__ void __launch_bounds__(256)
gemm_kernel(const float* __restrict__ deep,
            const float* __restrict__ shallow,
            const _Float16* __restrict__ projh,     // [n][k] f16
            const float* __restrict__ proj_b,
            const float* __restrict__ pos_emb,
            const int*   __restrict__ ws_topidx,
            float* __restrict__ out_tokens) {
  __shared__ __align__(16) _Float16 lA[2][32 * BSTR];   //  5 KB
  __shared__ __align__(16) _Float16 lB[2][D_ * BSTR];   // 40 KB
  __shared__ int s_rowoff[32];
  __shared__ int s_patch[32];

  const int tid  = threadIdx.x;
  const int wv   = tid >> 5;
  const int lane = tid & 31;
  const int mtile = wv & 1;
  const int ngrp  = wv >> 1;

  if (tid < 32) {
    const int gr = blockIdx.x * 32 + tid;
    const int b  = gr >> 6, ks = gr & 63;
    const int patch = ws_topidx[b * K_ + ks];
    s_patch[tid] = patch;
    const int ph = patch / NH_, pw = patch % NH_;
    s_rowoff[tid] = b * (C_ * HW_) + (ph * P_) * W_ + pw * P_;
  }
  __syncthreads();

  // ---- A staging: 128 threads; row base hoisted into registers ----
  const int a_row = tid >> 2;        // 0..31
  const int a_seg = tid & 3;         // 0..3 -> contiguous 8-float run
  const int a_base = (tid < 128) ? s_rowoff[a_row] : 0;
  const float* a_deep    = deep    + a_base;
  const float* a_shallow = shallow + a_base;
  _Float16* a_dst0 = &lA[0][a_row * BSTR + a_seg * 8];
  _Float16* a_dst1 = &lA[1][a_row * BSTR + a_seg * 8];

  auto stageA = [&](int s, int buf) {
    if (tid < 128) {
      const int k    = s * 32 + a_seg * 8;
      const int c    = k >> 6;
      const int pixi = k & 63;
      const int go   = c * HW_ + (pixi >> 3) * W_;
      const f32x4 d0 = *(const f32x4*)(a_deep    + go);
      const f32x4 d1 = *(const f32x4*)(a_deep    + go + 4);
      const f32x4 s0 = *(const f32x4*)(a_shallow + go);
      const f32x4 s1 = *(const f32x4*)(a_shallow + go + 4);
      v8h h;
#pragma unroll
      for (int i = 0; i < 4; ++i) h[i]     = (_Float16)(d0[i] + s0[i]);
#pragma unroll
      for (int i = 0; i < 4; ++i) h[4 + i] = (_Float16)(d1[i] + s1[i]);
      *(v8h*)(buf ? a_dst1 : a_dst0) = h;
    }
  };

  v8f acc[4];
#pragma unroll
  for (int j = 0; j < 4; ++j)
    acc[j] = (v8f){0.f, 0.f, 0.f, 0.f, 0.f, 0.f, 0.f, 0.f};

  const int g = lane >> 4;           // K-half selector (WMMA register layout)
  const int r = lane & 15;           // row (A) / column (B,D) within tile

  // ---- prologue: stage chunk 0 ----
  if (wv == 0) tdm_load_b((unsigned)(uintptr_t)&lB[0][0], projh);
  stageA(0, 0);

  for (int s = 0; s < NSTEP; ++s) {
    const int cur = s & 1;
    if (wv == 0) __builtin_amdgcn_s_wait_tensorcnt(0);
    __syncthreads();                 // chunk s (TDM B + DS A) visible to all.
                                     // Also: everyone has drained chunk s-1
                                     // reads, so buffers cur^1 are free.
    if (s + 1 < NSTEP) {             // overlap next chunk with compute
      if (wv == 0)
        tdm_load_b((unsigned)(uintptr_t)&lB[cur ^ 1][0],
                   projh + (size_t)(s + 1) * 32);
      stageA(s + 1, cur ^ 1);
    }

    // ---- A fragment: lane<16: K 0-7 & 16-23; lane>=16: K 8-15 & 24-31 ----
    const _Float16* ap = &lA[cur][(mtile * 16 + r) * BSTR];
    const v8h a0 = *(const v8h*)(ap + g * 8);
    const v8h a1 = *(const v8h*)(ap + 16 + g * 8);
    const v16h afrag = __builtin_shufflevector(a0, a1, 0, 1, 2, 3, 4, 5, 6, 7,
                                               8, 9, 10, 11, 12, 13, 14, 15);
#pragma unroll
    for (int j = 0; j < 4; ++j) {
      const int n = (ngrp * 4 + j) * 16 + r;       // B frag: k = g*8+i & 16+g*8+i, col n
      const _Float16* bp = &lB[cur][n * BSTR + g * 8];
      const v8h b0 = *(const v8h*)(bp);
      const v8h b1 = *(const v8h*)(bp + 16);
      const v16h bfrag = __builtin_shufflevector(b0, b1, 0, 1, 2, 3, 4, 5, 6, 7,
                                                 8, 9, 10, 11, 12, 13, 14, 15);
      acc[j] = __builtin_amdgcn_wmma_f32_16x16x32_f16(
          false, afrag, false, bfrag, (short)0, acc[j], false, false);
    }
    // no trailing barrier: double buffering + next iteration's top barrier
    // provide the required ordering.
  }

  // ---- epilogue: D rows = g*8 + rr, col = r (WMMA f32 C/D layout) ----
#pragma unroll
  for (int j = 0; j < 4; ++j) {
    const int n  = (ngrp * 4 + j) * 16 + r;
    const float pb = proj_b[n];
#pragma unroll
    for (int rr = 0; rr < 8; ++rr) {
      const int row = mtile * 16 + g * 8 + rr;
      const int gr  = blockIdx.x * 32 + row;
      const int patch = s_patch[row];
      out_tokens[(size_t)gr * D_ + n] =
          acc[j][rr] + pb + pos_emb[(size_t)patch * D_ + n];
    }
  }
}

// ---------------------------------------------------------------------------
extern "C" void kernel_launch(void* const* d_in, const int* in_sizes, int n_in,
                              void* d_out, int out_size, void* d_ws, size_t ws_size,
                              hipStream_t stream) {
  const float* deep    = (const float*)d_in[0];
  const float* shallow = (const float*)d_in[1];
  const float* sal_w   = (const float*)d_in[2];
  const float* sal_b   = (const float*)d_in[3];
  const float* proj_w  = (const float*)d_in[4];
  const float* proj_b  = (const float*)d_in[5];
  const float* pos_emb = (const float*)d_in[6];

  float* out        = (float*)d_out;
  float* out_tokens = out;                          // [32,64,256]
  float* out_idx    = out + B_ * K_ * D_;           // [32,64]
  float* out_chattn = out + B_ * K_ * D_ + B_ * K_; // [32,1]

  float* ws          = (float*)d_ws;
  float* ws_sal      = ws;                              // 401408 f
  float* ws_blocksum = ws_sal + (size_t)B_ * HW_;       // 1568 f
  float* ws_salflat  = ws_blocksum + B_ * 49;           // 6272 f
  int*   ws_topidx   = (int*)(ws_salflat + B_ * NP_);   // 2048 i32
  _Float16* ws_projh = (_Float16*)(ws_topidx + B_ * K_);// 16384*256 f16 (8 MB)

  sal_kernel    <<<B_ * 49, 256, 0, stream>>>(deep, sal_w, sal_b, ws_sal, ws_blocksum);
  pool_kernel   <<<B_,      256, 0, stream>>>(ws_sal, ws_blocksum, ws_salflat, out_chattn);
  topk_kernel   <<<B_,      256, 0, stream>>>(ws_salflat, ws_topidx, out_idx);
  convert_kernel<<<dim3(CPP_ / 32, D_ / 32), 256, 0, stream>>>(proj_w, ws_projh);
  gemm_kernel   <<<(B_ * K_) / 32, 256, 0, stream>>>(deep, shallow, ws_projh, proj_b,
                                                     pos_emb, ws_topidx, out_tokens);
}
